// QuantizedLinear_21663815041215
// MI455X (gfx1250) — compile-verified, matrix operand order checked by reading, not run on hardware
//
#include <hip/hip_runtime.h>
#include <stdint.h>

// ---- vector types (gfx1250, wave32) ----
typedef __attribute__((ext_vector_type(16))) _Float16 v16h;
typedef __attribute__((ext_vector_type(2)))  _Float16 h2;
typedef __attribute__((ext_vector_type(8)))  float    v8f;
typedef __attribute__((ext_vector_type(4)))  float    v4f;
typedef __attribute__((ext_vector_type(4)))  unsigned int v4u;

#define K_DIM 4096
#define N_DIM 12288

#define BM 128          // block tile M
#define BN 256          // block tile N
#define BK 64           // K chunk per stage (2 WMMA k-steps)
#define BKP (BK + 8)    // padded LDS row (f16 elems)
#define NCHUNK (K_DIM / BK)

#define A_TILE_ELEMS (BM * BKP)
#define W_TILE_ELEMS (BN * BKP)
#define SMEM_BYTES   (2 * (A_TILE_ELEMS + W_TILE_ELEMS) * 2)  // 110592 B

// ---- CDNA5 async global->LDS copy (16 B per lane), tracked by ASYNCcnt ----
__device__ __forceinline__ void async_cp16(const void* g, void* lds) {
    asm volatile("global_load_async_to_lds_b128 %0, %1, off"
                 :: "v"((uint32_t)(uintptr_t)lds),   // LDS byte address (low 32 bits of flat ptr)
                    "v"((uint64_t)(uintptr_t)g)      // 64-bit global address
                 : "memory");
}
__device__ __forceinline__ void wait_async0() {
    asm volatile("s_wait_asynccnt 0x0" ::: "memory");
}

// v_cvt_pk_rtz_f16_f32 (builtin returns __fp16 vec2; bit-cast to _Float16 vec2)
__device__ __forceinline__ h2 pkrtz(float a, float b) {
    return __builtin_bit_cast(h2, __builtin_amdgcn_cvt_pkrtz(a, b));
}

// ---- int8x4 -> 4 x fp16 (exact): u = w^0x80; fp16(0x6400|u) = 1024+u; -1152 => w
__device__ __forceinline__ void cvtW4(uint32_t w, h2& lo, h2& hi) {
    const h2 bias = {(_Float16)1152.0f, (_Float16)1152.0f};
    uint32_t x = w ^ 0x80808080u;
    lo = __builtin_bit_cast(h2, __builtin_amdgcn_perm(0x64646464u, x, 0x05010400u)) - bias;
    hi = __builtin_bit_cast(h2, __builtin_amdgcn_perm(0x64646464u, x, 0x04030402u)) - bias;
}

__device__ __forceinline__ v16h cvtW16(uint32_t w0, uint32_t w1, uint32_t w2, uint32_t w3) {
    union { h2 h[8]; v16h v; } u;
    cvtW4(w0, u.h[0], u.h[1]);
    cvtW4(w1, u.h[2], u.h[3]);
    cvtW4(w2, u.h[4], u.h[5]);
    cvtW4(w3, u.h[6], u.h[7]);
    return u.v;
}

__device__ __forceinline__ v16h cvtA16(v4f a, v4f b, v4f c, v4f d) {
    union { h2 h[8]; v16h v; } u;
    u.h[0] = pkrtz(a[0], a[1]);
    u.h[1] = pkrtz(a[2], a[3]);
    u.h[2] = pkrtz(b[0], b[1]);
    u.h[3] = pkrtz(b[2], b[3]);
    u.h[4] = pkrtz(c[0], c[1]);
    u.h[5] = pkrtz(c[2], c[3]);
    u.h[6] = pkrtz(d[0], d[1]);
    u.h[7] = pkrtz(d[2], d[3]);
    return u.v;
}

// ================= pre-pass converters (bandwidth bound) =================
__global__ __launch_bounds__(256)
void conv_a_f16(const float* __restrict__ in, _Float16* __restrict__ o) {
    size_t i = ((size_t)blockIdx.x * 256 + threadIdx.x) * 8;
    v4f a = *(const v4f*)(in + i);
    v4f b = *(const v4f*)(in + i + 4);
    union { h2 h[4]; v4u v; } u;
    u.h[0] = pkrtz(a[0], a[1]);
    u.h[1] = pkrtz(a[2], a[3]);
    u.h[2] = pkrtz(b[0], b[1]);
    u.h[3] = pkrtz(b[2], b[3]);
    *(v4u*)(o + i) = u.v;
}

__global__ __launch_bounds__(256)
void conv_w_f16(const v4u* __restrict__ in, _Float16* __restrict__ o) {
    size_t idx = (size_t)blockIdx.x * 256 + threadIdx.x;   // 16 int8 per thread
    v4u w = in[idx];
    *(v16h*)(o + idx * 16) = cvtW16(w[0], w[1], w[2], w[3]);
}

// ================= GEMM =================
template <bool PRE>
__global__ __launch_bounds__(256)
void qgemm_wmma_f16(const void* __restrict__ Ap, const void* __restrict__ Wp,
                    const float* __restrict__ scale, float* __restrict__ out) {
    extern __shared__ char smem[];
    _Float16* Abuf = (_Float16*)smem;                                         // [2][BM][BKP]
    _Float16* Wbuf = (_Float16*)(smem + 2 * A_TILE_ELEMS * sizeof(_Float16)); // [2][BN][BKP]

    const int tid  = threadIdx.x;
    const int lane = tid & 31;
    const int wave = tid >> 5;
    const int wm   = wave >> 2;     // 0..1 : 64-row slab
    const int wn   = wave & 3;      // 0..3 : 64-col slab

    const int m0 = blockIdx.y * BM;
    const int n0 = blockIdx.x * BN;

    // staging: A -> thread stages 32 elems of one row; W -> one full 64-elem row
    const int arow = tid >> 1;
    const int aseg = (tid & 1) * 32;

    const int rsel  = lane & 15;
    const int khalf = (lane >> 4) * 16;

    v8f acc[4][4] = {};

    // 2 k-steps x 16 WMMA from LDS buffer `cur`
    auto compute = [&](int cur) {
        const _Float16* Ab = Abuf + cur * A_TILE_ELEMS;
        const _Float16* Wb = Wbuf + cur * W_TILE_ELEMS;
        #pragma unroll
        for (int kk = 0; kk < 2; ++kk) {
            const int kbase = kk * 32 + khalf;
            v16h afrag[4], bfrag[4];
            #pragma unroll
            for (int mt = 0; mt < 4; ++mt)
                afrag[mt] = *(const v16h*)(Ab + (wm * 64 + mt * 16 + rsel) * BKP + kbase);
            #pragma unroll
            for (int nt = 0; nt < 4; ++nt)
                bfrag[nt] = *(const v16h*)(Wb + (wn * 64 + nt * 16 + rsel) * BKP + kbase);
            #pragma unroll
            for (int mt = 0; mt < 4; ++mt)
                #pragma unroll
                for (int nt = 0; nt < 4; ++nt)
                    acc[mt][nt] = __builtin_amdgcn_wmma_f32_16x16x32_f16(
                        false, afrag[mt], false, bfrag[nt],
                        (short)0, acc[mt][nt], false, false);
        }
    };

    if constexpr (PRE) {
        // -------- async staging: global -> LDS with no VGPR round-trip --------
        const _Float16* Af = (const _Float16*)Ap + (size_t)(m0 + arow) * K_DIM + aseg;
        const _Float16* Wf = (const _Float16*)Wp + (size_t)(n0 + tid) * K_DIM;

        auto issue_async = [&](int c, int buf) {
            const _Float16* ap = Af + c * BK;
            const _Float16* wp = Wf + c * BK;
            _Float16* An = Abuf + buf * A_TILE_ELEMS + arow * BKP + aseg;
            _Float16* Wn = Wbuf + buf * W_TILE_ELEMS + tid * BKP;
            #pragma unroll
            for (int i = 0; i < 4; ++i) async_cp16(ap + 8 * i, An + 8 * i);
            #pragma unroll
            for (int i = 0; i < 8; ++i) async_cp16(wp + 8 * i, Wn + 8 * i);
        };

        issue_async(0, 0);
        wait_async0();
        __syncthreads();

        for (int c = 0; c < NCHUNK; ++c) {
            const int cur = c & 1;
            if (c + 1 < NCHUNK) issue_async(c + 1, cur ^ 1);  // DMA overlaps WMMAs
            compute(cur);
            if (c + 1 < NCHUNK) {
                wait_async0();
                __syncthreads();
            }
        }
    } else {
        // -------- sync staging with in-register conversion --------
        const float*  A32 = (const float*)Ap + (size_t)(m0 + arow) * K_DIM + aseg;
        const int8_t* W8  = (const int8_t*)Wp + (size_t)(n0 + tid) * K_DIM;

        v4f rAf[8]; uint32_t rWi[16];
        auto issue = [&](int c) {
            const float*  ap = A32 + c * BK;
            const int8_t* wp = W8  + c * BK;
            #pragma unroll
            for (int i = 0; i < 8; ++i) rAf[i] = *(const v4f*)(ap + 4 * i);
            #pragma unroll
            for (int i = 0; i < 16; ++i) rWi[i] = *(const uint32_t*)(wp + 4 * i);
        };
        auto commit = [&](int buf) {
            _Float16* An = Abuf + buf * A_TILE_ELEMS + arow * BKP + aseg;
            _Float16* Wn = Wbuf + buf * W_TILE_ELEMS + tid * BKP;
            *(v16h*)(An)      = cvtA16(rAf[0], rAf[1], rAf[2], rAf[3]);
            *(v16h*)(An + 16) = cvtA16(rAf[4], rAf[5], rAf[6], rAf[7]);
            #pragma unroll
            for (int j = 0; j < 4; ++j)
                *(v16h*)(Wn + 16 * j) =
                    cvtW16(rWi[4*j], rWi[4*j+1], rWi[4*j+2], rWi[4*j+3]);
        };

        issue(0);
        commit(0);
        __syncthreads();

        for (int c = 0; c < NCHUNK; ++c) {
            const int cur = c & 1;
            if (c + 1 < NCHUNK) issue(c + 1);
            compute(cur);
            if (c + 1 < NCHUNK) {
                commit(cur ^ 1);
                __syncthreads();
            }
        }
    }

    // epilogue: per-channel scale, fp32 store
    // C/D layout: VGPR j -> M = j + 8*(lane/16), N = lane%16
    const int n_lo  = lane & 15;
    const int m_off = (lane >> 4) * 8;
    #pragma unroll
    for (int nt = 0; nt < 4; ++nt) {
        const int col = n0 + wn * 64 + nt * 16 + n_lo;
        const float s = scale[col];
        #pragma unroll
        for (int mt = 0; mt < 4; ++mt) {
            const int row = m0 + wm * 64 + mt * 16 + m_off;
            float* op = out + (size_t)row * N_DIM + col;
            #pragma unroll
            for (int j = 0; j < 8; ++j)
                op[(size_t)j * N_DIM] = acc[mt][nt][j] * s;
        }
    }
}

extern "C" void kernel_launch(void* const* d_in, const int* in_sizes, int n_in,
                              void* d_out, int out_size, void* d_ws, size_t ws_size,
                              hipStream_t stream) {
    const float*  inp   = (const float*)d_in[0];   // [M,4096] fp32 (M = 4096)
    const int8_t* qw    = (const int8_t*)d_in[1];  // [12288,4096] int8
    const float*  scale = (const float*)d_in[2];   // [12288] fp32
    float*        out   = (float*)d_out;

    const int M = in_sizes[0] / K_DIM;             // 4096
    const size_t needA = (size_t)M * K_DIM * sizeof(_Float16);       // 32 MB
    const size_t needW = (size_t)N_DIM * K_DIM * sizeof(_Float16);   // 96 MB

    dim3 grid(N_DIM / BN, M / BM);                 // 48 x 32

    if (ws_size >= needA + needW) {
        _Float16* Ah = (_Float16*)d_ws;
        _Float16* Wh = (_Float16*)((char*)d_ws + needA);
        conv_a_f16<<<dim3((unsigned)((size_t)M * K_DIM / (256 * 8))), dim3(256), 0, stream>>>(inp, Ah);
        conv_w_f16<<<dim3((unsigned)((size_t)N_DIM * K_DIM / (256 * 16))), dim3(256), 0, stream>>>(
            (const v4u*)qw, Wh);
        qgemm_wmma_f16<true><<<grid, dim3(256), SMEM_BYTES, stream>>>(Ah, Wh, scale, out);
    } else {
        qgemm_wmma_f16<false><<<grid, dim3(256), SMEM_BYTES, stream>>>(inp, qw, scale, out);
    }
}
